// AlignmentSimilarity_21088289423377
// MI455X (gfx1250) — matrix operand order repaired and checked by reading, not compiled, not run on hardware
//
#include <hip/hip_runtime.h>
#include <hip/hip_bf16.h>
#include <stdint.h>

// ---------------------------------------------------------------------------
// AlignmentSimilarity on MI455X (gfx1250): fused flash-attention style kernel.
//   sim = qc @ sc ; softmax ; aligned = sim @ sc^T   -- sim never materialized.
// CDNA5 paths used:
//   * v_wmma_f32_16x16x32_f16 (f16 in, f32 accumulate), wave32
//   * global_load_async_to_lds_b128 + s_wait_asynccnt double-buffered staging
// Inputs are pre-centered & pre-converted to f16 in workspace (two layouts)
// so every inner-loop global access is a contiguous async B128 -> LDS DMA.
// ---------------------------------------------------------------------------

typedef __attribute__((ext_vector_type(16))) _Float16 v16h;
typedef __attribute__((ext_vector_type(8)))  _Float16 v8h;
typedef __attribute__((ext_vector_type(8)))  float    v8f;

#define C_DIM 256
#define LPIX  1024
#define B_DIM 4
#define N_SUP 10
#define KSHOT 5
#define G_DIM 2            // N_SUP / KSHOT
#define QT    128          // query rows per block
#define TS    32           // s columns per inner step (one WMMA K)
#define NSTAGE (KSHOT * (LPIX / TS))   // 160 s-block stages per block

#define WAIT_ASYNCCNT(n) asm volatile("s_wait_asynccnt " n ::: "memory")

__device__ __forceinline__ void async_ld_b128(uint32_t lds_off, const void* g) {
  // VDST = LDS byte offset (AS3 offset = low 32 bits of generic pointer),
  // VADDR = 64-bit global address, SADDR = off  (GV mode). Tracked by ASYNCcnt.
  asm volatile("global_load_async_to_lds_b128 %0, %1, off"
               :: "v"(lds_off), "v"((unsigned long long)(uintptr_t)g)
               : "memory");
}

__device__ __forceinline__ v16h cat8(v8h lo, v8h hi) {
  return __builtin_shufflevector(lo, hi, 0,1,2,3,4,5,6,7,8,9,10,11,12,13,14,15);
}

__device__ __forceinline__ v8f wmma_f16(v16h a, v16h b, v8f c) {
  // D = A(16x32 f16) * B(32x16 f16) + C(16x16 f32)
  return __builtin_amdgcn_wmma_f32_16x16x32_f16(false, a, false, b, (short)0, c,
                                                false, false);
}

// ---------------- deterministic mean (two pass) ----------------
__global__ void partial_sum_kernel(const float* __restrict__ x, int n,
                                   float* __restrict__ part) {
  __shared__ float sm[8];
  float s = 0.0f;
  for (int i = blockIdx.x * blockDim.x + threadIdx.x; i < n;
       i += gridDim.x * blockDim.x)
    s += x[i];
  #pragma unroll
  for (int off = 16; off > 0; off >>= 1) s += __shfl_xor(s, off, 32);
  if ((threadIdx.x & 31) == 0) sm[threadIdx.x >> 5] = s;
  __syncthreads();
  if (threadIdx.x == 0) {
    float t = 0.0f;
    #pragma unroll
    for (int w = 0; w < 8; ++w) t += sm[w];
    part[blockIdx.x] = t;
  }
}

__global__ void final_mean_kernel(const float* __restrict__ part, int nb,
                                  float scale, float* __restrict__ dst) {
  if (threadIdx.x == 0) {
    float t = 0.0f;
    for (int i = 0; i < nb; ++i) t += part[i];   // fixed order -> deterministic
    *dst = t * scale;
  }
}

// ---------------- q_out: centered query replicated over class groups ------
__global__ void qout_kernel(const float* __restrict__ query,
                            const float* __restrict__ means,
                            float* __restrict__ out) {
  int i = blockIdx.x * blockDim.x + threadIdx.x;
  const int NQ = B_DIM * C_DIM * LPIX;
  if (i >= NQ) return;
  float v = query[i] - means[0];
  int b = i / (C_DIM * LPIX);
  int rest = i - b * (C_DIM * LPIX);
  out[(size_t)(b * G_DIM + 0) * (C_DIM * LPIX) + rest] = v;
  out[(size_t)(b * G_DIM + 1) * (C_DIM * LPIX) + rest] = v;
}

// ---------------- preprocess: center + f16, async-DMA-friendly layouts ----
__global__ void prep_query_kernel(const float* __restrict__ q,
                                  const float* __restrict__ means,
                                  _Float16* __restrict__ q16T) {  // [b][q][c]
  int i = blockIdx.x * blockDim.x + threadIdx.x;
  if (i >= B_DIM * C_DIM * LPIX) return;
  float v = q[i] - means[0];
  int b = i >> 18;               // / (C*L)
  int rem = i & ((C_DIM * LPIX) - 1);
  int c = rem >> 10;
  int qp = rem & (LPIX - 1);
  q16T[((size_t)b * LPIX + qp) * C_DIM + c] = (_Float16)v;
}

__global__ void prep_support_kernel(const float* __restrict__ s,
                                    const float* __restrict__ means,
                                    _Float16* __restrict__ s16,   // [n][s][c]
                                    _Float16* __restrict__ sT16) { // [n][c][s]
  int i = blockIdx.x * blockDim.x + threadIdx.x;
  if (i >= N_SUP * C_DIM * LPIX) return;
  float v = s[i] - means[1];
  _Float16 hv = (_Float16)v;
  sT16[i] = hv;
  int n = i >> 18;
  int rem = i & ((C_DIM * LPIX) - 1);
  int c = rem >> 10;
  int sp = rem & (LPIX - 1);
  s16[((size_t)n * LPIX + sp) * C_DIM + c] = hv;
}

// ---------------- fused attention + K-shot mean ---------------------------
__global__ __launch_bounds__(256, 1)
void flash_align_kernel(const _Float16* __restrict__ q16T,
                        const _Float16* __restrict__ s16,
                        const _Float16* __restrict__ sT16,
                        float* __restrict__ out) {
  // 64KB shared, aliased phases:
  //  phase 0: lds_q [QT][C] f16 (async-staged Q, consumed into VGPR frags)
  //  phase 1: two 32KB buffers, each: [TS][C] A-side @+0, [C][TS] B-side @+16KB
  //  phase 2: per-wave 16x16 f32 transpose scratch for coalesced stores
  __shared__ __align__(16) unsigned char smem[QT * C_DIM * 2];
  const uint32_t lds0 = (uint32_t)(uintptr_t)smem;   // AS3 byte offset

  const int tid  = threadIdx.x;
  const int wave = tid >> 5;
  const int lane = tid & 31;
  const int r    = lane & 15;
  const int h    = lane >> 4;

  const int b  = blockIdx.x;            // 0..3
  const int g  = blockIdx.y;            // 0..1 class group
  const int q0 = blockIdx.z * QT;       // query tile base

  // ---- async-stage Q tile (64KB contiguous) ----
  {
    const char* gQ = (const char*)(q16T + ((size_t)b * LPIX + q0) * C_DIM);
    #pragma unroll
    for (int i = 0; i < 16; ++i) {
      int idx = tid + i * 256;
      async_ld_b128(lds0 + idx * 16, gQ + idx * 16);
    }
  }
  WAIT_ASYNCCNT("0");
  __syncthreads();

  // ---- hoist q^T B-fragments into persistent VGPRs (8 x v16h) ----
  // B[k=c, n=q]: lane n=r, elem j -> c = cc*32 + 8h + j ; 8+j -> cc*32 + 16 + 8h + j
  v16h bq[C_DIM / 32];
  {
    _Float16 (*lds_q)[C_DIM] = (_Float16 (*)[C_DIM])smem;
    #pragma unroll
    for (int cc = 0; cc < C_DIM / 32; ++cc) {
      const v8h* row = (const v8h*)&lds_q[wave * 16 + r][cc * 32];
      bq[cc] = cat8(row[h], row[2 + h]);
    }
  }
  __syncthreads();   // everyone done reading Q before buffers alias its LDS

  // prefetch one stage: 8 async B128 per thread (4 A-side + 4 B-side)
  auto stage = [&](int it) {
    const int p  = it & 1;
    const int n  = g * KSHOT + (it >> 5);
    const int sb = (it & 31) * TS;
    // A-side: 16KB contiguous, s16[n][sb..sb+31][0..255]
    const char* gA = (const char*)(s16 + ((size_t)n * LPIX + sb) * C_DIM);
    const uint32_t lA = lds0 + (uint32_t)p * 32768u;
    #pragma unroll
    for (int i = 0; i < 4; ++i) {
      int idx = tid + i * 256;                      // 16B chunk id, 0..1023
      async_ld_b128(lA + idx * 16, gA + idx * 16);
    }
    // B-side: 64B per c-row, sT16[n][c][sb..sb+31]
    const char* gB = (const char*)(sT16 + ((size_t)n * C_DIM) * LPIX + sb);
    const uint32_t lB = lds0 + (uint32_t)p * 32768u + 16384u;
    #pragma unroll
    for (int i = 0; i < 4; ++i) {
      int idx = tid + i * 256;                      // c = idx>>2, chunk = idx&3
      int c = idx >> 2, ch = idx & 3;
      async_ld_b128(lB + c * 64 + ch * 16,
                    gB + (size_t)c * (LPIX * 2) + ch * 16);
    }
  };

  stage(0);

  v8f Oacc[16] = {};   // sum over shots of O_k / l_k
  v8f O[16] = {};
  float mrow = -__builtin_inff();
  float lrow = 0.0f;

  for (int it = 0; it < NSTAGE; ++it) {
    // prefetch next stage into the other buffer, then wait for current stage
    if (it + 1 < NSTAGE) { stage(it + 1); WAIT_ASYNCCNT("8"); }
    else                 { WAIT_ASYNCCNT("0"); }
    __syncthreads();     // whole buffer landed for every thread

    const int p = it & 1;
    _Float16 (*lds_s)[C_DIM] = (_Float16 (*)[C_DIM])(smem + p * 32768);
    _Float16 (*lds_sT)[TS]   = (_Float16 (*)[TS])(smem + p * 32768 + 16384);

    // ---- S^T = s_block * q^T : two 16x16 tiles (s rows 0-15 / 16-31) ----
    v8f st0 = {}, st1 = {};
    #pragma unroll
    for (int cc = 0; cc < C_DIM / 32; ++cc) {
      const v8h* r0 = (const v8h*)&lds_s[r][cc * 32];        // A lane = s row
      const v8h* r1 = (const v8h*)&lds_s[16 + r][cc * 32];
      v16h a0 = cat8(r0[h], r0[2 + h]);   // elems j: c=8h+j ; 8+j: c=16+8h+j
      v16h a1 = cat8(r1[h], r1[2 + h]);
      st0 = wmma_f16(a0, bq[cc], st0);
      st1 = wmma_f16(a1, bq[cc], st1);
    }
    // st0[j] = S[q=r, s=sb+8h+j], st1[j] = S[q=r, s=sb+16+8h+j]

    // ---- online softmax for this lane's query row r ----
    float mx = st0[0];
    #pragma unroll
    for (int j = 0; j < 8; ++j) { mx = fmaxf(mx, st0[j]); mx = fmaxf(mx, st1[j]); }
    mx = fmaxf(mx, __shfl_xor(mx, 16, 32));       // merge the two K-halves
    const float mnew  = fmaxf(mrow, mx);
    const float scale = __expf(mrow - mnew);      // first iter: exp(-inf)=0

    // P in f16, already in A-fragment layout (elem j <-> K=8h+j) in-lane
    v16h pfrag;
    float psum = 0.0f;
    #pragma unroll
    for (int j = 0; j < 8; ++j) {
      float e0 = __expf(st0[j] - mnew);
      float e1 = __expf(st1[j] - mnew);
      pfrag[j]     = (_Float16)e0;
      pfrag[8 + j] = (_Float16)e1;
      psum += e0 + e1;
    }
    psum += __shfl_xor(psum, 16, 32);
    lrow = lrow * scale + psum;
    mrow = mnew;

    // rescale O: element j belongs to q-row (j+8h); scale lives in lane h*8+j
    float sc[8];
    #pragma unroll
    for (int j = 0; j < 8; ++j) sc[j] = __shfl(scale, h * 8 + j, 32);
    #pragma unroll
    for (int t = 0; t < 16; ++t)
      #pragma unroll
      for (int j = 0; j < 8; ++j) O[t][j] *= sc[j];

    // ---- O += P * V : 16 c-tiles; B[k=s, n=c] from B-side buffer ----
    #pragma unroll
    for (int ct = 0; ct < 16; ++ct) {
      const v8h* rv = (const v8h*)&lds_sT[ct * 16 + r][0];
      v16h bv = cat8(rv[h], rv[2 + h]);  // elem j -> s = 8h + j ; 8+j -> s = 16 + 8h + j
      O[ct] = wmma_f16(pfrag, bv, O[ct]);
    }

    // ---- shot boundary: Oacc += O / l, reset softmax state ----
    if ((it & 31) == 31) {
      const float inv = 1.0f / lrow;
      float iv[8];
      #pragma unroll
      for (int j = 0; j < 8; ++j) iv[j] = __shfl(inv, h * 8 + j, 32);
      #pragma unroll
      for (int t = 0; t < 16; ++t) {
        #pragma unroll
        for (int j = 0; j < 8; ++j) Oacc[t][j] += O[t][j] * iv[j];
        O[t] = (v8f){};
      }
      mrow = -__builtin_inff();
      lrow = 0.0f;
    }
    __syncthreads();     // readers done before this buffer is prefetched again
  }

  // ---- write aligned output: per-wave 16x16 transpose for coalescing ----
  float (*scr)[16] = (float (*)[16])(smem + wave * (16 * 16 * 4));
  float* abase = out + (size_t)(B_DIM * G_DIM * C_DIM) * LPIX
                     + (size_t)(g * B_DIM + b) * C_DIM * LPIX;
  const float invK = 1.0f / (float)KSHOT;
  const int qg = q0 + wave * 16;

  #pragma unroll 1
  for (int ct = 0; ct < 16; ++ct) {
    #pragma unroll
    for (int j = 0; j < 8; ++j)
      scr[j + 8 * h][r] = Oacc[ct][j] * invK;      // scr[q_local][c_local]
    __syncthreads();                               // uniform; orders LDS RAW
    const int cl = lane >> 1;
    const int qh = (lane & 1) * 8;
    float* dst = abase + (size_t)(ct * 16 + cl) * LPIX + qg + qh;
    float4 v0 = make_float4(scr[qh + 0][cl], scr[qh + 1][cl],
                            scr[qh + 2][cl], scr[qh + 3][cl]);
    float4 v1 = make_float4(scr[qh + 4][cl], scr[qh + 5][cl],
                            scr[qh + 6][cl], scr[qh + 7][cl]);
    *(float4*)dst       = v0;
    *(float4*)(dst + 4) = v1;
    __syncthreads();                               // WAR before next tile
  }
}

extern "C" void kernel_launch(void* const* d_in, const int* in_sizes, int n_in,
                              void* d_out, int out_size, void* d_ws, size_t ws_size,
                              hipStream_t stream) {
  const float* q = (const float*)d_in[0];   // [4,256,32,32]
  const float* s = (const float*)d_in[1];   // [10,256,32,32]
  (void)in_sizes; (void)n_in; (void)out_size; (void)ws_size;
  float* out = (float*)d_out;
  float* ws  = (float*)d_ws;

  const int NQ = B_DIM * C_DIM * LPIX;      // 1,048,576
  const int NS = N_SUP * C_DIM * LPIX;      // 2,621,440

  // workspace byte layout (hipMalloc base is >=256B aligned):
  //   [0,4096)            : means (ws[0], ws[1]) + 512 reduction partials
  //   [4096, +2MB)        : q16T  f16 [b][q][c]
  //   next 5MB            : s16   f16 [n][s][c]
  //   next 5MB            : sT16  f16 [n][c][s]
  char* wsb = (char*)d_ws;
  _Float16* q16T = (_Float16*)(wsb + 4096);
  _Float16* s16  = (_Float16*)(wsb + 4096 + (size_t)NQ * 2);
  _Float16* sT16 = (_Float16*)(wsb + 4096 + (size_t)NQ * 2 + (size_t)NS * 2);

  partial_sum_kernel<<<256, 256, 0, stream>>>(q, NQ, ws + 2);
  final_mean_kernel<<<1, 32, 0, stream>>>(ws + 2, 256, 1.0f / (float)NQ, ws + 0);
  partial_sum_kernel<<<256, 256, 0, stream>>>(s, NS, ws + 2 + 256);
  final_mean_kernel<<<1, 32, 0, stream>>>(ws + 2 + 256, 256, 1.0f / (float)NS, ws + 1);

  prep_query_kernel<<<(NQ + 255) / 256, 256, 0, stream>>>(q, ws, q16T);
  prep_support_kernel<<<(NS + 255) / 256, 256, 0, stream>>>(s, ws, s16, sT16);

  qout_kernel<<<(NQ + 255) / 256, 256, 0, stream>>>(q, ws, out);

  flash_align_kernel<<<dim3(B_DIM, G_DIM, LPIX / QT), 256, 0, stream>>>(
      q16T, s16, sT16, out);
}
